// OutfitCompatibilityModel_31387620999443
// MI455X (gfx1250) — matrix operand order repaired and checked by reading, not compiled, not run on hardware
//
#include <hip/hip_runtime.h>
#include <hip/hip_fp16.h>
#include <math.h>

typedef __attribute__((ext_vector_type(16))) _Float16 v16h;
typedef __attribute__((ext_vector_type(8)))  float    v8f;
typedef __attribute__((ext_vector_type(4)))  float    v4f;

#define PROJ    1000
#define NPAD    1024
#define KDIM    96
#define NPAIR   15
#define BT      16          // batch rows per workgroup
#define ROWS    (BT*5)      // 80 GEMM rows per workgroup
#define NCHUNK  128
#define NCH     (NPAD/NCHUNK)
#define FSTRIDE 132         // padded LDS row stride (words); 132*4=528 ≡ 0 mod 16

// ---------------- prep kernels ----------------

// W_proj (96 x 1000 f32) -> W16t (1024 x 96 f16, transposed, zero-padded in N)
__global__ void prep_w(const float* __restrict__ Wp, _Float16* __restrict__ W16t) {
  int idx = blockIdx.x * 256 + threadIdx.x;         // over KDIM*NPAD
  if (idx >= KDIM * NPAD) return;
  int k = idx / NPAD;
  int n = idx % NPAD;
  float v = (n < PROJ) ? Wp[k * PROJ + n] : 0.0f;
  W16t[(size_t)n * KDIM + k] = (_Float16)v;
}

// fold b_proj + BN1 into per-column scale/shift; mw2 = relu(masks_w)^2 (padded)
__global__ void prep_ab(const float* __restrict__ bp,
                        const float* __restrict__ g,  const float* __restrict__ b,
                        const float* __restrict__ m,  const float* __restrict__ v,
                        const float* __restrict__ masksw,
                        float* __restrict__ alpha, float* __restrict__ beta,
                        float* __restrict__ mw2) {
  int idx = blockIdx.x * 256 + threadIdx.x;         // over 16*NPAD
  int r = idx / NPAD, j = idx % NPAD;
  if (r == 0) {
    float a = 0.f, bb = 0.f;
    if (j < PROJ) {
      a  = g[j] * rsqrtf(v[j] + 1e-5f);
      bb = (bp[j] - m[j]) * a + b[j];
    }
    alpha[j] = a;
    beta[j]  = bb;
  } else if (r <= NPAIR) {
    int p = r - 1;
    float w = 0.f;
    if (j < PROJ) w = fmaxf(masksw[p * PROJ + j], 0.f);
    mw2[p * NPAD + j] = w * w;
  }
}

// ---------------- fused GEMM + BN + ReLU + pairwise cosine ----------------

__device__ __forceinline__ float group16_sum(float x) {
  x += __shfl_xor(x, 1, 32);
  x += __shfl_xor(x, 2, 32);
  x += __shfl_xor(x, 4, 32);
  x += __shfl_xor(x, 8, 32);
  return x;
}

__global__ __launch_bounds__(256) void fused_pair(
    const float* __restrict__ emb, const unsigned char* __restrict__ mask,
    const _Float16* __restrict__ W16t,
    const float* __restrict__ alpha, const float* __restrict__ beta,
    const float* __restrict__ mw2,
    const float* __restrict__ bnr_g, const float* __restrict__ bnr_b,
    const float* __restrict__ bnr_m, const float* __restrict__ bnr_v,
    float* __restrict__ rel, int B)
{
  __shared__ float feat_s[ROWS * FSTRIDE];   // 42240 B
  __shared__ float mw2_s[NPAIR * NCHUNK];    //  7680 B

  constexpr int IIc[NPAIR] = {0,0,0,0,0,1,1,1,1,2,2,2,3,3,4};
  constexpr int JJc[NPAIR] = {0,1,2,3,4,1,2,3,4,2,3,4,3,4,4};

  const int t    = threadIdx.x;
  const int wave = t >> 5;
  const int lane = t & 31;
  const int hl   = lane & 15;
  const int hi   = lane >> 4;          // 0 or 1 (half-wave)
  const int wg   = blockIdx.x;
  const int row_base = wg * ROWS;
  const int max_row  = B * 5 - 1;
  const int bl = t >> 4;               // local batch 0..15
  const int s  = t & 15;               // 8-wide column slab within the 128 chunk

  // accumulators: 15 dots; norms only needed for the 10 off-diagonal pairs
  // (diagonal pairs have ni == nj == dot); guarded accumulators get DCE'd.
  float dotA[NPAIR], niA[NPAIR], njA[NPAIR];
#pragma unroll
  for (int p = 0; p < NPAIR; ++p) { dotA[p] = 0.f; niA[p] = 0.f; njA[p] = 0.f; }

  const int colA0 = wave * 16 + hl;    // C-fragment column within the 128 chunk

  for (int c = 0; c < NCH; ++c) {
    const int nbase = c * NCHUNK;

    // stage mw2 chunk into LDS
    for (int idx = t; idx < NPAIR * NCHUNK; idx += 256)
      mw2_s[idx] = mw2[(idx / NCHUNK) * NPAD + nbase + (idx & (NCHUNK - 1))];

    // ---- phase 1: WMMA GEMM chunk + BN + ReLU into LDS ----
    const int colg = nbase + colA0;    // global column < NPAD
    v16h bf[3];
#pragma unroll
    for (int k = 0; k < 3; ++k) {
      int k0 = k * 32 + (hi ? 16 : 0); // B layout: 16 contiguous K per lane
      bf[k] = *(const v16h*)(W16t + (size_t)colg * KDIM + k0);
    }
    const float al = alpha[colg];
    const float be = beta[colg];

#pragma unroll
    for (int m = 0; m < 5; ++m) {
      v8f acc = {0.f, 0.f, 0.f, 0.f, 0.f, 0.f, 0.f, 0.f};
      int rowg = row_base + m * 16 + hl;
      if (rowg > max_row) rowg = max_row;  // clamp (writes guarded later)
#pragma unroll
      for (int k = 0; k < 3; ++k) {
        int k0 = k * 32 + (hi ? 16 : 0);   // A layout: 16 contiguous K per lane
        const float* ap = emb + (size_t)rowg * KDIM + k0;
        v4f a0 = *(const v4f*)(ap);
        v4f a1 = *(const v4f*)(ap + 4);
        v4f a2 = *(const v4f*)(ap + 8);
        v4f a3 = *(const v4f*)(ap + 12);
        v16h af;
#pragma unroll
        for (int e = 0; e < 4; ++e) {
          af[e]      = (_Float16)a0[e];
          af[4 + e]  = (_Float16)a1[e];
          af[8 + e]  = (_Float16)a2[e];
          af[12 + e] = (_Float16)a3[e];
        }
        acc = __builtin_amdgcn_wmma_f32_16x16x32_f16(
            false, af, false, bf[k], (short)0, acc, false, false);
      }
#pragma unroll
      for (int v = 0; v < 8; ++v) {
        int rloc = m * 16 + v + (hi ? 8 : 0);
        feat_s[rloc * FSTRIDE + colA0] = fmaxf(al * acc[v] + be, 0.f);
      }
    }
    __syncthreads();

    // ---- phase 2: accumulate pairwise dots and masked norms ----
    // thread (bl, s) owns contiguous columns [s*8, s*8+8) -> all LDS reads b128
#pragma unroll
    for (int h = 0; h < 2; ++h) {
      const int n0 = s * 8 + h * 4;
      v4f f[5];
#pragma unroll
      for (int i = 0; i < 5; ++i)
        f[i] = *(const v4f*)(feat_s + (bl * 5 + i) * FSTRIDE + n0);
      v4f w2v[NPAIR];
#pragma unroll
      for (int p = 0; p < NPAIR; ++p)
        w2v[p] = *(const v4f*)(mw2_s + p * NCHUNK + n0);
#pragma unroll
      for (int e = 0; e < 4; ++e) {
#pragma unroll
        for (int p = 0; p < NPAIR; ++p) {
          const float w2 = w2v[p][e];
          const float fi = f[IIc[p]][e], fj = f[JJc[p]][e];
          dotA[p] = fmaf(fi * fj, w2, dotA[p]);
          if (IIc[p] != JJc[p]) {        // compile-time: diagonal norms == dot
            niA[p] = fmaf(fi * fi, w2, niA[p]);
            njA[p] = fmaf(fj * fj, w2, njA[p]);
          }
        }
      }
    }
    __syncthreads();
  }

  // ---- reduce across the 16 column-slabs and emit BN'd relations ----
  const int b = wg * BT + bl;
#pragma unroll
  for (int p = 0; p < NPAIR; ++p) {
    const float d = group16_sum(dotA[p]);
    float na, nb;
    if (IIc[p] == JJc[p]) { na = d; nb = d; }
    else { na = group16_sum(niA[p]); nb = group16_sum(njA[p]); }
    if (s == p && b < B) {
      float denom = fmaxf(sqrtf(na), 1e-12f) * fmaxf(sqrtf(nb), 1e-12f);
      float r = d / denom;
      bool valid = mask[b * 5 + IIc[p]] && mask[b * 5 + JJc[p]];
      r = valid ? r : 0.f;
      float o = (r - bnr_m[p]) * rsqrtf(bnr_v[p] + 1e-5f) * bnr_g[p] + bnr_b[p];
      rel[b * NPAIR + p] = o;
    }
  }
}

// ---------------- tiny MLP: (B,15) -> (B,) ----------------

__global__ __launch_bounds__(256) void mlp_head(
    const float* __restrict__ rel,
    const float* __restrict__ W1, const float* __restrict__ b1,
    const float* __restrict__ g2, const float* __restrict__ bb2,
    const float* __restrict__ m2, const float* __restrict__ v2,
    const float* __restrict__ W2, const float* __restrict__ b2,
    const float* __restrict__ g3, const float* __restrict__ bb3,
    const float* __restrict__ m3, const float* __restrict__ v3,
    const float* __restrict__ W3, const float* __restrict__ b3,
    float* __restrict__ out)
{
  __shared__ float sr[NPAIR];
  __shared__ float h1[256];
  __shared__ float h2[64];
  const int b = blockIdx.x, t = threadIdx.x;

  if (t < NPAIR) sr[t] = rel[b * NPAIR + t];
  __syncthreads();

  float a = b1[t];
#pragma unroll
  for (int k = 0; k < NPAIR; ++k) a = fmaf(sr[k], W1[k * 256 + t], a);
  a = (a - m2[t]) * rsqrtf(v2[t] + 1e-5f) * g2[t] + bb2[t];
  h1[t] = fmaxf(a, 0.f);
  __syncthreads();

  if (t < 64) {
    float a2 = b2[t];
    for (int k = 0; k < 256; ++k) a2 = fmaf(h1[k], W2[k * 64 + t], a2);
    a2 = (a2 - m3[t]) * rsqrtf(v3[t] + 1e-5f) * g3[t] + bb3[t];
    h2[t] = fmaxf(a2, 0.f);
  }
  __syncthreads();

  if (t == 0) {
    float a3 = b3[0];
#pragma unroll
    for (int k = 0; k < 64; ++k) a3 = fmaf(h2[k], W3[k], a3);
    out[b] = 1.f / (1.f + expf(-a3));
  }
}

// ---------------- launch ----------------

extern "C" void kernel_launch(void* const* d_in, const int* in_sizes, int n_in,
                              void* d_out, int out_size, void* d_ws, size_t ws_size,
                              hipStream_t stream) {
  const float*         emb    = (const float*)d_in[0];
  const unsigned char* mask   = (const unsigned char*)d_in[1];
  const float* Wp   = (const float*)d_in[2];
  const float* bp   = (const float*)d_in[3];
  const float* bn1g = (const float*)d_in[4];
  const float* bn1b = (const float*)d_in[5];
  const float* bn1m = (const float*)d_in[6];
  const float* bn1v = (const float*)d_in[7];
  const float* mskw = (const float*)d_in[8];
  const float* bnrg = (const float*)d_in[9];
  const float* bnrb = (const float*)d_in[10];
  const float* bnrm = (const float*)d_in[11];
  const float* bnrv = (const float*)d_in[12];
  const float* W1   = (const float*)d_in[13];
  const float* b1   = (const float*)d_in[14];
  const float* g2   = (const float*)d_in[15];
  const float* bb2  = (const float*)d_in[16];
  const float* m2   = (const float*)d_in[17];
  const float* v2   = (const float*)d_in[18];
  const float* W2   = (const float*)d_in[19];
  const float* b2   = (const float*)d_in[20];
  const float* g3   = (const float*)d_in[21];
  const float* bb3  = (const float*)d_in[22];
  const float* m3   = (const float*)d_in[23];
  const float* v3   = (const float*)d_in[24];
  const float* W3   = (const float*)d_in[25];
  const float* b3   = (const float*)d_in[26];

  const int B = in_sizes[0] / (5 * KDIM);   // 8192

  // workspace layout (all 256B-aligned offsets)
  char* ws = (char*)d_ws;
  _Float16* W16t  = (_Float16*)(ws);                       // 1024*96*2   = 196608
  float*    alpha = (float*)(ws + 196608);                 // 1024*4
  float*    beta  = (float*)(ws + 200704);                 // 1024*4
  float*    mw2   = (float*)(ws + 204800);                 // 15*1024*4   =  61440
  float*    rel   = (float*)(ws + 266240);                 // B*15*4

  prep_w<<<(KDIM * NPAD + 255) / 256, 256, 0, stream>>>(Wp, W16t);
  prep_ab<<<(16 * NPAD + 255) / 256, 256, 0, stream>>>(
      bp, bn1g, bn1b, bn1m, bn1v, mskw, alpha, beta, mw2);

  fused_pair<<<(B + BT - 1) / BT, 256, 0, stream>>>(
      emb, mask, W16t, alpha, beta, mw2, bnrg, bnrb, bnrm, bnrv, rel, B);

  mlp_head<<<B, 256, 0, stream>>>(
      rel, W1, b1, g2, bb2, m2, v2, W2, b2, g3, bb3, m3, v3, W3, b3,
      (float*)d_out);
}
